// CrossAttention3D_84885733638497
// MI455X (gfx1250) — matrix operand order repaired and checked by reading, not compiled, not run on hardware
//
#include <hip/hip_runtime.h>

// ---------------------------------------------------------------------------
// CrossAttention3D for MI455X (gfx1250): bf16 WMMA pipeline.
// B=1, C=512, T=4, H=W=32  ->  N = M = 4096, 8 heads, hd=64, groups=8.
// ---------------------------------------------------------------------------

typedef __attribute__((ext_vector_type(16))) __bf16 v16bf;
typedef __attribute__((ext_vector_type(8)))  float  v8f;
typedef __attribute__((ext_vector_type(4)))  int    v4i;

#define CC    512
#define NN    4096
#define NHEAD 8
#define HD    64
#define KPAD  72   // LDS row stride (elements): 144B -> conflict-free b128 reads

#if defined(__HIP_DEVICE_COMPILE__) && \
    __has_builtin(__builtin_amdgcn_global_load_async_to_lds_b128)
#define HAVE_ASYNC_LDS 1
#else
#define HAVE_ASYNC_LDS 0
#endif

union FragU { v16bf v; uint4 q[2]; };

// Build a 16x(bf16) WMMA fragment from two 16-byte chunks (2x b128 loads).
static __device__ inline v16bf ldfrag(const __bf16* p0, const __bf16* p1) {
  FragU u;
  u.q[0] = *(const uint4*)p0;
  u.q[1] = *(const uint4*)p1;
  return u.v;
}

static __device__ inline v8f vzero() {
  v8f z = {0.f, 0.f, 0.f, 0.f, 0.f, 0.f, 0.f, 0.f};
  return z;
}

static __device__ inline v8f wmma_bf16(v16bf a, v16bf b, v8f c) {
  return __builtin_amdgcn_wmma_f32_16x16x32_bf16(false, a, false, b,
                                                 (short)0, c, false, false);
}

static __device__ inline float fast_exp2(float x) {
#if defined(__HIP_DEVICE_COMPILE__) && __has_builtin(__builtin_amdgcn_exp2f)
  return __builtin_amdgcn_exp2f(x);
#else
  return exp2f(x);
#endif
}

// xor-shuffle within each 16-lane half: VALU v_permlane16 when available.
static __device__ inline float lane_xor(float x, unsigned s0, unsigned s1,
                                        int mask) {
#if defined(__HIP_DEVICE_COMPILE__) && __has_builtin(__builtin_amdgcn_permlane16)
  unsigned u = __float_as_uint(x);
  u = __builtin_amdgcn_permlane16(u, u, s0, s1, false, false);
  (void)mask;
  return __uint_as_float(u);
#else
  return __shfl_xor(x, mask, 32);
#endif
}

// Reduce across the 16-lane half (row-major WMMA D layout).
static __device__ inline float red16_max(float r) {
  r = fmaxf(r, lane_xor(r, 0x67452301u, 0xEFCDAB89u, 1));
  r = fmaxf(r, lane_xor(r, 0x54761032u, 0xDCFE98BAu, 2));
  r = fmaxf(r, lane_xor(r, 0x32107654u, 0xBA98FEDCu, 4));
  r = fmaxf(r, lane_xor(r, 0xFEDCBA98u, 0x76543210u, 8));
  return r;
}
static __device__ inline float red16_add(float r) {
  r += lane_xor(r, 0x67452301u, 0xEFCDAB89u, 1);
  r += lane_xor(r, 0x54761032u, 0xDCFE98BAu, 2);
  r += lane_xor(r, 0x32107654u, 0xBA98FEDCu, 4);
  r += lane_xor(r, 0xFEDCBA98u, 0x76543210u, 8);
  return r;
}

// Stage 16 bytes global -> LDS (async-to-LDS on CDNA5, fallback ld+ds_store).
static __device__ inline void stage16(__bf16* lds, const __bf16* g) {
#if HAVE_ASYNC_LDS
  __builtin_amdgcn_global_load_async_to_lds_b128(
      (__attribute__((address_space(1))) v4i*)g,
      (__attribute__((address_space(3))) v4i*)lds, 0, 0);
#else
  *(uint4*)lds = *(const uint4*)g;
#endif
}

static __device__ inline void stage_wait() {
#if HAVE_ASYNC_LDS
#if __has_builtin(__builtin_amdgcn_s_wait_asynccnt)
  __builtin_amdgcn_s_wait_asynccnt(0);
#else
  asm volatile("s_wait_asynccnt 0x0" ::: "memory");
#endif
#endif
}

// ---------------------------------------------------------------------------
// Kernel 1: per-group mean / rstd.  16 blocks: [tensor(2)][group(8)].
// ---------------------------------------------------------------------------
__global__ void gn_stats_kernel(const float* __restrict__ x,
                                const float* __restrict__ ctx,
                                float* __restrict__ stats) {
  const int bid = blockIdx.x;
  const int t = bid >> 3, g = bid & 7;
  const float4* src = (const float4*)((t ? ctx : x) + (size_t)g * 64 * NN);
  float s = 0.f, s2 = 0.f;
  for (int i = threadIdx.x; i < 65536; i += blockDim.x) {
    float4 v = src[i];
    s  += v.x + v.y + v.z + v.w;
    s2 += v.x * v.x + v.y * v.y + v.z * v.z + v.w * v.w;
  }
  __shared__ float ra[256], rb[256];
  ra[threadIdx.x] = s;
  rb[threadIdx.x] = s2;
  __syncthreads();
  for (int st = 128; st > 0; st >>= 1) {
    if ((int)threadIdx.x < st) {
      ra[threadIdx.x] += ra[threadIdx.x + st];
      rb[threadIdx.x] += rb[threadIdx.x + st];
    }
    __syncthreads();
  }
  if (threadIdx.x == 0) {
    const float inv = 1.f / 262144.f;
    float mu  = ra[0] * inv;
    float var = rb[0] * inv - mu * mu;
    stats[bid * 2 + 0] = mu;
    stats[bid * 2 + 1] = rsqrtf(var + 1e-5f);
  }
}

// ---------------------------------------------------------------------------
// Kernel 2: groupnorm apply, write TRANSPOSED bf16 activations Xt[n][c].
// ---------------------------------------------------------------------------
__global__ void gn_apply_kernel(const float* __restrict__ src,
                                const float* __restrict__ stats,
                                const float* __restrict__ w,
                                const float* __restrict__ b,
                                __bf16* __restrict__ dst_t) {
  const int idx = blockIdx.x * 256 + threadIdx.x;   // 512*4096 exactly
  const int c = idx & (CC - 1);
  const int n = idx >> 9;
  const int g = c >> 6;
  const float mu = stats[g * 2 + 0];
  const float rs = stats[g * 2 + 1];
  const float v = src[(size_t)c * NN + n];
  dst_t[idx] = (__bf16)(((v - mu) * rs) * w[c] + b[c]);
}

__global__ void cvt_bf16_kernel(const float* __restrict__ s,
                                __bf16* __restrict__ d, int n) {
  const int i = blockIdx.x * 256 + threadIdx.x;
  if (i < n) d[i] = (__bf16)s[i];
}

// ---------------------------------------------------------------------------
// Kernel 4: WMMA GEMM  Y[o][n] = sum_c W[o][c] * Xt[n][c] + bias[o]
// MODE 0: q -> qt[h][n][d] bf16 * oscale (scale*log2e folded in)
// MODE 1: kv -> kt[h][m][d] bf16 + v natural v[d][m] bf16
// MODE 2: proj -> f32 + residual
// ---------------------------------------------------------------------------
template <int MODE>
__global__ __launch_bounds__(32)
void gemm_wmma_kernel(const __bf16* __restrict__ W,
                      const __bf16* __restrict__ Xt,
                      const float* __restrict__ bias,
                      void* __restrict__ outp,
                      const float* __restrict__ resid,
                      float oscale) {
  const int lane = threadIdx.x;
  const int li = lane & 15, lh = lane >> 4;
  const int kb = lh * 8;     // A-fragment K base
  const int kb2 = lh * 16;   // unused; B fragments also use kb
  const int oTile = blockIdx.y * 64;
  const int nTile = blockIdx.x * 32;

  v8f acc[4][2];
#pragma unroll
  for (int so = 0; so < 4; ++so)
#pragma unroll
    for (int sn = 0; sn < 2; ++sn) acc[so][sn] = vzero();

  for (int cB = 0; cB < CC; cB += 32) {
    v16bf a[4], bf[2];
#pragma unroll
    for (int so = 0; so < 4; ++so) {
      const __bf16* wp = W + (size_t)(oTile + so * 16 + li) * CC + cB;
      a[so] = ldfrag(wp + kb, wp + 16 + kb);
    }
#pragma unroll
    for (int sn = 0; sn < 2; ++sn) {
      const __bf16* xp = Xt + (size_t)(nTile + sn * 16 + li) * CC + cB + kb;
      bf[sn] = ldfrag(xp, xp + 16);
    }
#pragma unroll
    for (int so = 0; so < 4; ++so)
#pragma unroll
      for (int sn = 0; sn < 2; ++sn)
        acc[so][sn] = wmma_bf16(a[so], bf[sn], acc[so][sn]);
  }

#pragma unroll
  for (int so = 0; so < 4; ++so)
#pragma unroll
    for (int sn = 0; sn < 2; ++sn)
#pragma unroll
      for (int v = 0; v < 8; ++v) {
        const int o = oTile + so * 16 + v + 8 * lh;
        const int n = nTile + sn * 16 + li;
        float val = acc[so][sn][v] + bias[o];
        if (MODE == 0) {
          ((__bf16*)outp)[((size_t)(o >> 6) * NN + n) * HD + (o & 63)] =
              (__bf16)(val * oscale);
        } else if (MODE == 1) {
          if (o < 512)
            ((__bf16*)outp)[((size_t)(o >> 6) * NN + n) * HD + (o & 63)] =
                (__bf16)val;
          else
            (((__bf16*)outp) + (size_t)512 * NN)[(size_t)(o - 512) * NN + n] =
                (__bf16)val;
        } else {
          ((float*)outp)[(size_t)o * NN + n] = val + resid[(size_t)o * NN + n];
        }
      }
}

// ---------------------------------------------------------------------------
// Kernel 5: flash attention.  4 waves / block, 64 query rows; K/V 64x64 tiles
// cooperatively staged in LDS (async-to-LDS); online softmax in exp2 domain,
// permlane16 row-max, per-lane deferred row-sums.
// ---------------------------------------------------------------------------
__global__ __launch_bounds__(128)
void attn_kernel(const __bf16* __restrict__ qt,
                 const __bf16* __restrict__ kt,
                 const __bf16* __restrict__ vn,
                 __bf16* __restrict__ att_t) {
  const int tid = threadIdx.x;
  const int lane = tid & 31, wv = tid >> 5;
  const int li = lane & 15, lh = lane >> 4;
  const int kb = lh * 8, kb2 = lh * 16;
  const int h = blockIdx.y;
  const int nBase = blockIdx.x * 64 + wv * 16;

  __shared__ alignas(16) __bf16 Ks[64 * KPAD];      // K tile [m][d]
  __shared__ alignas(16) __bf16 Vs[64 * KPAD];      // V tile [d][m]
  __shared__ alignas(16) __bf16 Ps[4][16 * KPAD];   // per-wave P transpose

  // Q fragments (16 rows x 64 d); scale*log2e already folded in.
  const __bf16* qrow = qt + ((size_t)h * NN + nBase + li) * HD;
  const v16bf qa0 = ldfrag(qrow + kb, qrow + 16 + kb);        // d = 0..31
  const v16bf qa1 = ldfrag(qrow + 32 + kb, qrow + 48 + kb);   // d = 32..63

  float mrun[8], psum[8];
  v8f Oacc[4];
#pragma unroll
  for (int v = 0; v < 8; ++v) { mrun[v] = -3.0e38f; psum[v] = 0.f; }
#pragma unroll
  for (int dt = 0; dt < 4; ++dt) Oacc[dt] = vzero();

  const __bf16* kgBase = kt + (size_t)h * NN * HD;
  const __bf16* vgBase = vn + (size_t)h * HD * NN;

  for (int mt = 0; mt < 64; ++mt) {
    const int mBase = mt * 64;

    // ---- cooperative staging of K (64x64) and V (64x64) tiles ----
    {
      const __bf16* kg = kgBase + (size_t)mBase * HD;
      const __bf16* vg = vgBase + mBase;
#pragma unroll
      for (int i = 0; i < 4; ++i) {
        const int c = tid + 128 * i;               // 512 chunks of 16B
        const int r = c >> 3, col = (c & 7) * 8;
        stage16(Ks + r * KPAD + col, kg + (size_t)r * HD + col);
        stage16(Vs + r * KPAD + col, vg + (size_t)r * NN + col);
      }
    }
    stage_wait();
    __syncthreads();

    // ---- S = Q K^T : 16 x 64 score strip (4 D tiles) ----
    v8f s[4];
#pragma unroll
    for (int t = 0; t < 4; ++t) {
      const __bf16* krow = Ks + (t * 16 + li) * KPAD + kb;
      s[t] = wmma_bf16(qa0, ldfrag(krow, krow + 16), vzero());
      s[t] = wmma_bf16(qa1, ldfrag(krow + 32, krow + 48), s[t]);
    }

    // ---- online softmax (exp2 domain), rows = v + 8*lh in 16-lane halves --
    float fac[8];
#pragma unroll
    for (int v = 0; v < 8; ++v) {
      float r = fmaxf(fmaxf(s[0][v], s[1][v]), fmaxf(s[2][v], s[3][v]));
      r = red16_max(r);
      const float mnew = fmaxf(mrun[v], r);
      fac[v] = fast_exp2(mrun[v] - mnew);
      mrun[v] = mnew;
      const float p0 = fast_exp2(s[0][v] - mnew);
      const float p1 = fast_exp2(s[1][v] - mnew);
      const float p2 = fast_exp2(s[2][v] - mnew);
      const float p3 = fast_exp2(s[3][v] - mnew);
      psum[v] = psum[v] * fac[v] + (p0 + p1 + p2 + p3);  // per-lane partial
      __bf16* prow = Ps[wv] + (v + 8 * lh) * KPAD;
      prow[li]      = (__bf16)p0;
      prow[16 + li] = (__bf16)p1;
      prow[32 + li] = (__bf16)p2;
      prow[48 + li] = (__bf16)p3;
    }

    // P as A-fragments (same-wave LDS ops are in-order; no barrier needed)
    const __bf16* prow = Ps[wv] + li * KPAD;
    const v16bf pa0 = ldfrag(prow + kb, prow + 16 + kb);        // m 0..31
    const v16bf pa1 = ldfrag(prow + 32 + kb, prow + 48 + kb);   // m 32..63

    // ---- O = O*fac + P V ----
#pragma unroll
    for (int dt = 0; dt < 4; ++dt) {
      const __bf16* vrow = Vs + (dt * 16 + li) * KPAD + kb;
      const v16bf vb0 = ldfrag(vrow, vrow + 16);
      const v16bf vb1 = ldfrag(vrow + 32, vrow + 48);
#pragma unroll
      for (int v = 0; v < 8; ++v) Oacc[dt][v] *= fac[v];
      Oacc[dt] = wmma_bf16(pa0, vb0, Oacc[dt]);
      Oacc[dt] = wmma_bf16(pa1, vb1, Oacc[dt]);
    }
    __syncthreads();   // protect Ks/Vs before next staging round
  }

  float inv[8];
#pragma unroll
  for (int v = 0; v < 8; ++v) inv[v] = 1.f / red16_add(psum[v]);
#pragma unroll
  for (int dt = 0; dt < 4; ++dt)
#pragma unroll
    for (int v = 0; v < 8; ++v) {
      const int n = nBase + v + 8 * lh;
      att_t[(size_t)n * CC + h * HD + dt * 16 + li] =
          (__bf16)(Oacc[dt][v] * inv[v]);
    }
}

// ---------------------------------------------------------------------------
// Host launcher
// ---------------------------------------------------------------------------
extern "C" void kernel_launch(void* const* d_in, const int* in_sizes, int n_in,
                              void* d_out, int out_size, void* d_ws,
                              size_t ws_size, hipStream_t stream) {
  const float* x    = (const float*)d_in[0];
  const float* ctx  = (const float*)d_in[1];
  const float* nqw  = (const float*)d_in[2];
  const float* nqb  = (const float*)d_in[3];
  const float* nkw  = (const float*)d_in[4];
  const float* nkb  = (const float*)d_in[5];
  const float* qw   = (const float*)d_in[6];
  const float* qb   = (const float*)d_in[7];
  const float* kvw  = (const float*)d_in[8];
  const float* kvb  = (const float*)d_in[9];
  const float* pw   = (const float*)d_in[10];
  const float* pb   = (const float*)d_in[11];
  (void)in_sizes; (void)n_in; (void)out_size; (void)ws_size;

  char* ws = (char*)d_ws;
  float*  stats = (float*)ws;
  __bf16* xn_t  = (__bf16*)(ws + 256);
  __bf16* cn_t  = xn_t + (size_t)CC * NN;
  __bf16* q_t   = cn_t + (size_t)CC * NN;
  __bf16* kvbuf = q_t  + (size_t)CC * NN;
  __bf16* k_t   = kvbuf;
  __bf16* v_n   = kvbuf + (size_t)CC * NN;
  __bf16* att_t = v_n   + (size_t)CC * NN;
  __bf16* wq_b  = att_t + (size_t)CC * NN;
  __bf16* wkv_b = wq_b  + (size_t)CC * CC;
  __bf16* wp_b  = wkv_b + (size_t)2 * CC * CC;

  // hd^-0.5 * log2(e): softmax runs in exp2 domain.
  const float qscale = 0.125f * 1.44269504088896340736f;

  gn_stats_kernel<<<16, 256, 0, stream>>>(x, ctx, stats);
  gn_apply_kernel<<<(CC * NN) / 256, 256, 0, stream>>>(x, stats, nqw, nqb,
                                                       xn_t);
  gn_apply_kernel<<<(CC * NN) / 256, 256, 0, stream>>>(ctx, stats + 16, nkw,
                                                       nkb, cn_t);
  cvt_bf16_kernel<<<(CC * CC) / 256, 256, 0, stream>>>(qw, wq_b, CC * CC);
  cvt_bf16_kernel<<<(2 * CC * CC) / 256, 256, 0, stream>>>(kvw, wkv_b,
                                                           2 * CC * CC);
  cvt_bf16_kernel<<<(CC * CC) / 256, 256, 0, stream>>>(pw, wp_b, CC * CC);

  gemm_wmma_kernel<0><<<dim3(NN / 32, CC / 64), 32, 0, stream>>>(
      wq_b, xn_t, qb, (void*)q_t, nullptr, qscale);
  gemm_wmma_kernel<1><<<dim3(NN / 32, (2 * CC) / 64), 32, 0, stream>>>(
      wkv_b, cn_t, kvb, (void*)kvbuf, nullptr, 1.0f);

  attn_kernel<<<dim3(NN / 64, NHEAD), 128, 0, stream>>>(q_t, k_t, v_n, att_t);

  gemm_wmma_kernel<2><<<dim3(128, 8), 32, 0, stream>>>(
      wp_b, att_t, pb, d_out, x, 1.0f);
}
